// PixelVAE_18313740550485
// MI455X (gfx1250) — matrix operand order repaired and checked by reading, not compiled, not run on hardware
//
#include <hip/hip_runtime.h>

// ---------------------------------------------------------------------------
// PixelVAE forward on MI455X (gfx1250, wave32, WMMA + Tensor Data Mover).
// Decoder convs = implicit GEMM on v_wmma_f32_16x16x32_f16 (f32 accumulate).
// Activations NHWC f16. Weights pre-converted once to f16 [tap][cout][cin];
// per-block staging of the input slab and double-buffered per-tap weight
// tiles goes through TENSOR_LOAD_TO_LDS (TDM) + s_wait_tensorcnt.
// ---------------------------------------------------------------------------

typedef __attribute__((ext_vector_type(16))) _Float16 v16h;
typedef __attribute__((ext_vector_type(8)))  _Float16 v8h;
typedef __attribute__((ext_vector_type(8)))  float    v8f;
typedef __attribute__((ext_vector_type(4)))  float    v4f;
typedef __attribute__((ext_vector_type(4)))  unsigned int v4u;
typedef __attribute__((ext_vector_type(8)))  int      v8i;
typedef __attribute__((ext_vector_type(4)))  int      v4i;

#if defined(__has_builtin)
#  if __has_builtin(__builtin_amdgcn_tensor_load_to_lds)
#    define PVAE_TDM 1
#  endif
#endif
#ifndef PVAE_TDM
#  define PVAE_TDM 0
#endif

#if PVAE_TDM
// 1-D TDM copy: `bytes` (multiple of 8, 8B-aligned src/dst) global -> LDS.
// D# packed per CDNA5 ISA 8.3/8.4: count=1, type=2, data_size=3 (8B units).
__device__ __forceinline__ void tdm_load_1d(unsigned int lds_addr,
                                            const void* gptr, unsigned int bytes) {
  unsigned long long ga = (unsigned long long)gptr;
  unsigned int n8 = bytes >> 3;                       // 8-byte units
  v4u g0 = { 1u,                                      // count=1 (valid user D#)
             lds_addr,                                 // lds_addr [63:32]
             (unsigned int)ga,                         // global_addr [95:64]
             ((unsigned int)(ga >> 32) & 0x01FFFFFFu)  // global_addr [120:96]
               | 0x80000000u };                        // type=2 -> bits[127:126]=10b
  v8i g1 = { (int)(3u << 16),                          // mask=0, data_size=3 (8B)
             (int)((n8 & 0xFFFFu) << 16),              // tensor_dim0 lo16 @ [63:48]
             (int)(((n8 >> 16) & 0xFFFFu) | (1u << 16)), // td0 hi16, tensor_dim1=1
             (int)((n8 & 0xFFFFu) << 16),              // td1 hi=0, tile_dim0 = n8
             1,                                        // tile_dim1=1, tile_dim2=0
             (int)n8,                                  // tensor_dim0_stride lo32
             0, 0 };                                   // stride hi / dim1_stride
  v4i g2 = { 0, 0, 0, 0 };
  v4i g3 = { 0, 0, 0, 0 };
#if defined(__clang_major__) && (__clang_major__ >= 23)
  v8i gz = { 0, 0, 0, 0, 0, 0, 0, 0 };
  __builtin_amdgcn_tensor_load_to_lds(g0, g1, g2, g3, gz, 0);
#else
  __builtin_amdgcn_tensor_load_to_lds(g0, g1, g2, g3, 0);
#endif
}
#endif

// ---------------- encoder conv1: (B,1,32,32)->(B,16,16,64) NHWC f16 --------
__global__ void k_enc_conv1(const int* __restrict__ x, const float* __restrict__ w,
                            const float* __restrict__ bias, _Float16* __restrict__ h1) {
  int pix = blockIdx.x;            // B*16*16 = 32768
  int co  = threadIdx.x;           // 64
  int b = pix >> 8, p = pix & 255, y = p >> 4, xx = p & 15;
  float acc = bias[co];
#pragma unroll
  for (int ky = 0; ky < 3; ++ky) {
    int sy = 2 * y - 1 + ky;
    if (sy < 0 || sy >= 32) continue;
#pragma unroll
    for (int kx = 0; kx < 3; ++kx) {
      int sx = 2 * xx - 1 + kx;
      if (sx < 0 || sx >= 32) continue;
      float v = 2.0f * (float)x[(b << 10) + (sy << 5) + sx] - 1.0f;
      acc += v * w[co * 9 + ky * 3 + kx];
    }
  }
  acc = acc > 0.f ? acc : 0.2f * acc;   // leaky_relu 0.2
  h1[((size_t)pix << 6) + co] = (_Float16)acc;
}

// ------------- encoder conv2: (B,16,16,64) -> h2 (B, 4096) NCHW-flat f16 ---
__global__ void k_enc_conv2(const _Float16* __restrict__ h1, const float* __restrict__ w,
                            const float* __restrict__ bias, _Float16* __restrict__ h2) {
  int b = blockIdx.x >> 6, co = blockIdx.x & 63;     // grid B*64
  int y = threadIdx.x >> 3, xx = threadIdx.x & 7;    // block 64
  float acc = bias[co];
  for (int ky = 0; ky < 3; ++ky) {
    int sy = 2 * y - 1 + ky;
    if (sy < 0 || sy >= 16) continue;
    for (int kx = 0; kx < 3; ++kx) {
      int sx = 2 * xx - 1 + kx;
      if (sx < 0 || sx >= 16) continue;
      const _Float16* ip = h1 + (((size_t)(b * 16 + sy) * 16 + sx) << 6);
      const float* wp = w + (size_t)co * 576 + ky * 3 + kx;  // stride 9 per cin
#pragma unroll 8
      for (int ci = 0; ci < 64; ++ci)
        acc += (float)ip[ci] * wp[ci * 9];
    }
  }
  acc = acc > 0.f ? acc : 0.2f * acc;
  // NCHW-flat feature index so it matches enc_fcw column order
  h2[(size_t)b * 4096 + co * 64 + y * 8 + xx] = (_Float16)acc;
}

// ------------- encoder FC via WMMA: (128 x 4096) @ (4096 x 512) ------------
__global__ void k_fc_wmma(const _Float16* __restrict__ h2, const float* __restrict__ fcw,
                          const float* __restrict__ fcb, float* __restrict__ ep) {
  int mt = blockIdx.x & 7, nt = blockIdx.x >> 3;   // grid 256, block 32 (1 wave)
  int lane = threadIdx.x;
  int half = (lane >= 16) ? 1 : 0;
  int mr = lane & 15;
  v8f acc = {};
  int m = mt * 16 + mr;
  int n = nt * 16 + mr;
  const _Float16* arow = h2 + (size_t)m * 4096;
  const float*    brow = fcw + (size_t)n * 4096;
  for (int s = 0; s < 128; ++s) {
    int ka = s * 32 + (half ? 16 : 0);
    v8h lo = *(const v8h*)(arow + ka);
    v8h hi = *(const v8h*)(arow + ka + 8);
    v16h a;
#pragma unroll
    for (int e = 0; e < 8; ++e) { a[e] = lo[e]; a[e + 8] = hi[e]; }
    int kb = s * 32 + (half ? 16 : 0);
    __builtin_prefetch(brow + kb + 256, 0, 1);   // global_prefetch_b8 next chunk
    v4f f0 = *(const v4f*)(brow + kb);
    v4f f1 = *(const v4f*)(brow + kb + 4);
    v4f f2 = *(const v4f*)(brow + kb + 8);
    v4f f3 = *(const v4f*)(brow + kb + 12);
    v16h bf;
#pragma unroll
    for (int e = 0; e < 4; ++e) {
      bf[e]      = (_Float16)f0[e];
      bf[e + 4]  = (_Float16)f1[e];
      bf[e + 8]  = (_Float16)f2[e];
      bf[e + 12] = (_Float16)f3[e];
    }
    acc = __builtin_amdgcn_wmma_f32_16x16x32_f16(false, a, false, bf,
                                                 (short)0, acc, false, false);
  }
#pragma unroll
  for (int v = 0; v < 8; ++v) {
    int mm = mt * 16 + v + (half ? 8 : 0);
    ep[(size_t)mm * 512 + n] = acc[v] + fcb[n];
  }
}

// ------------- reparam + per-image KL partials -----------------------------
__global__ void k_reparam_kl(const float* __restrict__ ep, const float* __restrict__ eps,
                             float* __restrict__ z, float* __restrict__ kl_part) {
  __shared__ float red[256];
  int b = blockIdx.x, j = threadIdx.x;      // grid 128, block 256
  float mu = ep[b * 512 + j];
  float ls = ep[b * 512 + 256 + j];
  z[b * 256 + j] = mu + expf(ls) * eps[b * 256 + j];
  red[j] = 0.5f * (mu * mu + expf(2.f * ls) - 2.f * ls - 1.f);
  __syncthreads();
  for (int st = 128; st > 0; st >>= 1) {
    if (j < st) red[j] += red[j + st];
    __syncthreads();
  }
  if (j == 0) kl_part[b] = red[0];
}

// ------------- conditioning linears (tiny) ---------------------------------
__global__ void k_cond(const float* __restrict__ z,
                       const float* w0, const float* w1, const float* w2,
                       const float* w3, const float* w4,
                       const float* b0, const float* b1, const float* b2,
                       const float* b3, const float* b4,
                       float* __restrict__ cond) {
  int i = blockIdx.x >> 7, b = blockIdx.x & 127, n = threadIdx.x;  // grid 640, blk 64
  const float* w  = (i == 0) ? w0 : (i == 1) ? w1 : (i == 2) ? w2 : (i == 3) ? w3 : w4;
  const float* bb = (i == 0) ? b0 : (i == 1) ? b1 : (i == 2) ? b2 : (i == 3) ? b3 : b4;
  int outc = (i == 4) ? 2 : 64;
  float acc = 0.f;
  if (n < outc) {
    acc = bb[n];
    const float* wr = w + n * 256;
    const float* zr = z + b * 256;
    for (int k = 0; k < 256; ++k) acc += zr[k] * wr[k];
  }
  cond[((size_t)i * 128 + b) * 64 + n] = acc;   // zero-padded past outc
}

// ------------- decoder weight prep: OIHW f32 -> [tap][cout][cin] f16 -------
// Masked taps of the 7x7 kernel are exactly linear indices 0..24 (ky*7+kx).
__global__ void k_prep_w(const float* __restrict__ w, _Float16* __restrict__ out,
                         int outc_valid, int outc_pad) {
  int e = blockIdx.x * 256 + threadIdx.x;
  int total = 25 * outc_pad * 64;
  if (e >= total) return;
  int cin = e & 63;
  int n = (e >> 6) % outc_pad;
  int t = e / (outc_pad * 64);                 // 0..24 == ky*7+kx
  float v = 0.f;
  if (n < outc_valid) v = w[((size_t)(n * 64 + cin)) * 49 + t];
  out[e] = (_Float16)v;
}

// ------------- decoder layer 0 (mask A, cin=1) direct ----------------------
__global__ void k_dec0(const int* __restrict__ x, const float* __restrict__ w,
                       const float* __restrict__ bias, const float* __restrict__ cond,
                       _Float16* __restrict__ out) {
  size_t g = (size_t)blockIdx.x * 256 + threadIdx.x;   // grid 32768
  int n = (int)(g & 63);
  int pix = (int)(g >> 6);
  int b = pix >> 10, p = pix & 1023, y = p >> 5, xx = p & 31;
  float acc = bias[n] + cond[b * 64 + n];
  for (int ky = 0; ky < 4; ++ky) {
    int sy = y + ky - 3;
    if (sy < 0) continue;
    int kxmax = (ky < 3) ? 7 : 3;            // mask 'A': strictly left of center
    for (int kx = 0; kx < kxmax; ++kx) {
      int sx = xx + kx - 3;
      if (sx < 0 || sx >= 32) continue;
      float v = 2.f * (float)x[(b << 10) + (sy << 5) + sx] - 1.f;
      acc += v * w[n * 49 + ky * 7 + kx];
    }
  }
  acc = acc > 0.f ? acc : 0.f;
  out[g] = (_Float16)acc;                    // NHWC: pix*64 + n == g
}

// ------------- decoder masked conv (mask B) via WMMA implicit GEMM ---------
// 256 threads = 8 waves; wave owns 16 pixels x NTILES*16 channels. Input slab
// (7 rows x 32 px x 64 ch, contiguous in NHWC) staged to LDS once; per-tap
// weight tiles double-buffered via TDM while WMMAs run.
template <int NTILES, bool RELU, bool LAST>
__global__ __launch_bounds__(256)
void k_dec_wmma(const _Float16* __restrict__ in, const _Float16* __restrict__ wf,
                const float* __restrict__ bias, const float* __restrict__ cond,
                _Float16* __restrict__ out_h, float* __restrict__ out_logits) {
  constexpr int OUTC  = NTILES * 16;
  constexpr int WTILE = OUTC * 64;               // halves per tap tile
  __shared__ __align__(32) _Float16 lds_in[7 * 32 * 64];
  __shared__ __align__(32) _Float16 lds_w[2][WTILE];

  int lane = threadIdx.x & 31;
  int wave = threadIdx.x >> 5;
  int half = (lane >= 16) ? 1 : 0;
  int mr = lane & 15;
  int pix0 = blockIdx.x << 7;                    // 128 px/block, grid 1024
  int b = pix0 >> 10;
  int y0 = (pix0 & 1023) >> 5;                   // 4 rows per block
  int y  = y0 + (wave >> 1);
  int x0 = (wave & 1) << 4;
  int pixBase = pix0 + (wave << 4);

  int srcrow = (y0 >= 3) ? (y0 - 3) : 0;
  int dstrow = srcrow - (y0 - 3);                // 3 when y0==0, else 0
  int nrows  = 7 - dstrow;
  const _Float16* slab_src = in + (((size_t)b * 1024 + srcrow * 32) << 6);

#if PVAE_TDM
  if (wave == 0) {
    tdm_load_1d((unsigned int)(unsigned long long)(lds_in + dstrow * 2048),
                slab_src, (unsigned int)(nrows * 4096));
    tdm_load_1d((unsigned int)(unsigned long long)(&lds_w[0][0]),
                wf, (unsigned int)(WTILE * 2));
    __builtin_amdgcn_s_wait_tensorcnt(0);
  }
#else
  {
    const unsigned int* gs = (const unsigned int*)slab_src;
    unsigned int* ld = (unsigned int*)(lds_in + dstrow * 2048);
    for (int i = threadIdx.x; i < nrows * 1024; i += 256) ld[i] = gs[i];
    const unsigned int* gw = (const unsigned int*)wf;
    unsigned int* lw = (unsigned int*)&lds_w[0][0];
    for (int i = threadIdx.x; i < WTILE / 2; i += 256) lw[i] = gw[i];
  }
#endif
  __syncthreads();

  v8f acc[NTILES] = {};
  for (int t = 0; t < 25; ++t) {
#if PVAE_TDM
    if (wave == 0) {
      if (t + 1 < 25) {                          // prefetch next tap's weights
        tdm_load_1d((unsigned int)(unsigned long long)(&lds_w[(t + 1) & 1][0]),
                    wf + (size_t)(t + 1) * WTILE, (unsigned int)(WTILE * 2));
        __builtin_amdgcn_s_wait_tensorcnt(1);    // tap t's copy complete
      } else {
        __builtin_amdgcn_s_wait_tensorcnt(0);
      }
    }
#else
    if (t + 1 < 25) {
      const unsigned int* gw = (const unsigned int*)(wf + (size_t)(t + 1) * WTILE);
      unsigned int* lw = (unsigned int*)&lds_w[(t + 1) & 1][0];
      for (int i = threadIdx.x; i < WTILE / 2; i += 256) lw[i] = gw[i];
    }
#endif
    __syncthreads();                             // tap t weights visible

    int ky = t / 7, kx = t % 7;                  // t<25: (3,0..3) for 21..24
    int sy = y + ky - 3;
    int sx = x0 + mr + kx - 3;
    bool valid = (sy >= 0) && (sx >= 0) && (sx < 32);
    int lrow = sy - (y0 - 3);                    // 0..6 when valid
    const _Float16* wbuf = &lds_w[t & 1][0];
#pragma unroll
    for (int s = 0; s < 2; ++s) {                // K = 64 cin = 2 x 32
      v16h a = {};
      if (valid) {
        const _Float16* ap = lds_in + (((lrow << 5) + sx) << 6) + s * 32 + (half ? 16 : 0);
        v8h lo = *(const v8h*)ap;
        v8h hi = *(const v8h*)(ap + 8);
#pragma unroll
        for (int e = 0; e < 8; ++e) { a[e] = lo[e]; a[e + 8] = hi[e]; }
      }
#pragma unroll
      for (int tt = 0; tt < NTILES; ++tt) {
        v16h bf = *(const v16h*)&wbuf[((tt * 16 + mr) << 6) + s * 32 + (half ? 16 : 0)];
        acc[tt] = __builtin_amdgcn_wmma_f32_16x16x32_f16(false, a, false, bf,
                                                         (short)0, acc[tt], false, false);
      }
    }
    __syncthreads();                             // done with lds_w[t&1]
  }

  // epilogue: + bias + cond(z), optional relu, store
  int cb = b * 64;
#pragma unroll
  for (int t = 0; t < NTILES; ++t) {
    int n = t * 16 + mr;
    float bc;
    if (LAST) bc = (n < 2) ? (bias[n] + cond[cb + n]) : 0.f;
    else      bc = bias[n] + cond[cb + n];
#pragma unroll
    for (int v = 0; v < 8; ++v) {
      int mrow = v + (half ? 8 : 0);
      float val = acc[t][v] + bc;
      if (RELU) val = val > 0.f ? val : 0.f;
      if (LAST) {
        if (n < 2) out_logits[(size_t)(pixBase + mrow) * 2 + n] = val;
      } else {
        out_h[((size_t)(pixBase + mrow) << 6) + n] = (_Float16)val;
      }
    }
  }
}

// ------------- recon loss partials (deterministic 2-stage reduction) -------
__global__ void k_loss_recon(const float* __restrict__ logits, const int* __restrict__ x,
                             float* __restrict__ part) {
  __shared__ float red[256];
  int g = blockIdx.x * 256 + threadIdx.x;    // 131072 pixels, grid 512
  float l0 = logits[(size_t)g * 2];
  float l1 = logits[(size_t)g * 2 + 1];
  int t = x[g];
  float m = fmaxf(l0, l1);
  float lse = m + logf(expf(l0 - m) + expf(l1 - m));
  red[threadIdx.x] = lse - (t ? l1 : l0);
  __syncthreads();
  for (int st = 128; st > 0; st >>= 1) {
    if (threadIdx.x < st) red[threadIdx.x] += red[threadIdx.x + st];
    __syncthreads();
  }
  if (threadIdx.x == 0) part[blockIdx.x] = red[0];
}

__global__ void k_finalize(const float* __restrict__ recon_part,
                           const float* __restrict__ kl_part, float* __restrict__ out) {
  if (threadIdx.x == 0 && blockIdx.x == 0) {
    float s = 0.f;
    for (int i = 0; i < 512; ++i) s += recon_part[i];
    float k = 0.f;
    for (int i = 0; i < 128; ++i) k += kl_part[i];
    out[0] = (s + k) / 128.f;                // recon: sum/B ; kl: mean over B
  }
}

// ---------------------------------------------------------------------------
extern "C" void kernel_launch(void* const* d_in, const int* in_sizes, int n_in,
                              void* d_out, int out_size, void* d_ws, size_t ws_size,
                              hipStream_t stream) {
  (void)in_sizes; (void)n_in; (void)out_size; (void)ws_size;
  const int*   x      = (const int*)  d_in[0];
  const float* eps    = (const float*)d_in[1];
  const float* enc_w1 = (const float*)d_in[2];
  const float* enc_b1 = (const float*)d_in[3];
  const float* enc_w2 = (const float*)d_in[4];
  const float* enc_b2 = (const float*)d_in[5];
  const float* fcw    = (const float*)d_in[6];
  const float* fcb    = (const float*)d_in[7];
  const float *dw[5], *db[5], *cw[5], *cb[5];
  for (int i = 0; i < 5; ++i) {
    dw[i] = (const float*)d_in[8 + i];
    db[i] = (const float*)d_in[13 + i];
    cw[i] = (const float*)d_in[18 + i];
    cb[i] = (const float*)d_in[23 + i];
  }

  char* ws = (char*)d_ws;
  size_t off = 0;
  auto take = [&](size_t bytes) {
    char* ptr = ws + off;
    off = (off + bytes + 255) & ~(size_t)255;
    return ptr;
  };
  float*    recon_part = (float*)   take(512 * 4);
  float*    kl_part    = (float*)   take(128 * 4);
  _Float16* h1         = (_Float16*)take((size_t)128 * 16 * 16 * 64 * 2);
  _Float16* h2         = (_Float16*)take((size_t)128 * 4096 * 2);
  float*    epar       = (float*)   take((size_t)128 * 512 * 4);
  float*    z          = (float*)   take((size_t)128 * 256 * 4);
  float*    cond       = (float*)   take((size_t)5 * 128 * 64 * 4);
  _Float16* actA       = (_Float16*)take((size_t)128 * 1024 * 64 * 2);
  _Float16* actB       = (_Float16*)take((size_t)128 * 1024 * 64 * 2);
  float*    logits     = (float*)   take((size_t)128 * 1024 * 2 * 4);
  _Float16* wf1        = (_Float16*)take((size_t)25 * 64 * 64 * 2);
  _Float16* wf2        = (_Float16*)take((size_t)25 * 64 * 64 * 2);
  _Float16* wf3        = (_Float16*)take((size_t)25 * 64 * 64 * 2);
  _Float16* wf4        = (_Float16*)take((size_t)25 * 16 * 64 * 2);

  // weight prep (f32 OIHW -> f16 [tap][cout][cin], masked taps only)
  k_prep_w<<<400, 256, 0, stream>>>(dw[1], wf1, 64, 64);
  k_prep_w<<<400, 256, 0, stream>>>(dw[2], wf2, 64, 64);
  k_prep_w<<<400, 256, 0, stream>>>(dw[3], wf3, 64, 64);
  k_prep_w<<<100, 256, 0, stream>>>(dw[4], wf4, 2, 16);

  // encoder
  k_enc_conv1<<<32768, 64, 0, stream>>>(x, enc_w1, enc_b1, h1);
  k_enc_conv2<<<8192, 64, 0, stream>>>(h1, enc_w2, enc_b2, h2);
  k_fc_wmma<<<256, 32, 0, stream>>>(h2, fcw, fcb, epar);
  k_reparam_kl<<<128, 256, 0, stream>>>(epar, eps, z, kl_part);
  k_cond<<<640, 64, 0, stream>>>(z, cw[0], cw[1], cw[2], cw[3], cw[4],
                                 cb[0], cb[1], cb[2], cb[3], cb[4], cond);
  // decoder
  k_dec0<<<32768, 256, 0, stream>>>(x, dw[0], db[0], cond + 0 * 128 * 64, actA);
  k_dec_wmma<4, true,  false><<<1024, 256, 0, stream>>>(actA, wf1, db[1], cond + 1 * 128 * 64, actB, nullptr);
  k_dec_wmma<4, true,  false><<<1024, 256, 0, stream>>>(actB, wf2, db[2], cond + 2 * 128 * 64, actA, nullptr);
  k_dec_wmma<4, true,  false><<<1024, 256, 0, stream>>>(actA, wf3, db[3], cond + 3 * 128 * 64, actB, nullptr);
  k_dec_wmma<1, false, true ><<<1024, 256, 0, stream>>>(actB, wf4, db[4], cond + 4 * 128 * 64, nullptr, logits);
  // losses
  k_loss_recon<<<512, 256, 0, stream>>>(logits, x, recon_part);
  k_finalize<<<1, 64, 0, stream>>>(recon_part, kl_part, (float*)d_out);
}
